// TPMultiHeadAttn_20435454394534
// MI455X (gfx1250) — compile-verified, matrix operand order checked by reading, not run on hardware
//
#include <hip/hip_runtime.h>

// ---------------------------------------------------------------------------
// MI455X (gfx1250) TP multi-head attention:
//   RMSNorm -> QKV GEMM (WMMA f16, async-LDS staged B) -> flash attention
//   (WMMA f16, f32 online softmax) -> output GEMM (WMMA f16) + bias.
// CDNA5 paths: v_wmma_f32_16x16x32_f16, global_load_async_to_lds_b128
// (ASYNCcnt), ds_load_b128 fragment reads, s_wait_asynccnt/s_wait_dscnt,
// sched_group_barrier-pipelined DS-read -> WMMA groups.
// ---------------------------------------------------------------------------

typedef _Float16 f16;
typedef __attribute__((ext_vector_type(16))) _Float16 v16h;
typedef __attribute__((ext_vector_type(8)))  _Float16 v8h;
typedef __attribute__((ext_vector_type(8)))  float    v8f;

#define B_    2
#define S_    2048
#define D_    1024
#define H_    16
#define DH_   64
#define NTOK  (B_ * S_)          // 4096
#define HE_   (H_ * DH_)         // 1024
#define LDB_  40                 // padded LDS row stride (halves): 80B = 5*16B

static __device__ __forceinline__ v8f wmma16(v16h a, v16h b, v8f c) {
  // D = A(16x32 f16) * B(32x16 f16) + C(16x16 f32)
  return __builtin_amdgcn_wmma_f32_16x16x32_f16(
      /*neg_a=*/false, a, /*neg_b=*/false, b,
      /*c_mod=*/(short)0, c, /*reuse_a=*/false, /*reuse_b=*/false);
}

// Per-lane fragment load for the CDNA5 16-bit A layout (and, from an
// N-major [n][k] array, the matching B layout):
//   lane%16 selects the row (A) / column (B); hs = lane>>4 selects the
//   K half: elements j<8 -> k = hs*8+j ; j>=8 -> k = 16+hs*8+(j-8).
// p points at element [lane%16-row][k0]; both 8-half loads are 16B aligned.
static __device__ __forceinline__ v16h frag_ld(const f16* p, int hs) {
  v8h lo = *(const v8h*)(p + hs * 8);
  v8h hi = *(const v8h*)(p + 16 + hs * 8);
  return __builtin_shufflevector(lo, hi,
                                 0, 1, 2, 3, 4, 5, 6, 7,
                                 8, 9, 10, 11, 12, 13, 14, 15);
}

// CDNA5 async copy: 16B global -> LDS per lane, tracked by ASYNCcnt.
static __device__ __forceinline__ void async_b128(const f16* g, unsigned lds_off) {
  asm volatile("global_load_async_to_lds_b128 %0, %1, off"
               :: "v"(lds_off), "v"(g) : "memory");
}
static __device__ __forceinline__ void wait_async0() {
  asm volatile("s_wait_asynccnt 0x0" ::: "memory");
}

// Stage a 64x32 f16 B tile (N-major source, ld = 1024) into LDS with padded
// rows (LDB_ halves).  128 threads x 2 chunks x 16B = 4 KB.
static __device__ __forceinline__ void stage_b64x32(const f16* Wtile, f16* lbuf,
                                                    int tid) {
  #pragma unroll
  for (int i = 0; i < 2; ++i) {
    const int ch  = tid + i * 128;        // 0..255
    const int row = ch >> 2, c4 = ch & 3; // 4x16B chunks per 64B row
    const f16* g = Wtile + (size_t)row * 1024 + c4 * 8;
    const unsigned l = (unsigned)(uintptr_t)(lbuf + row * LDB_ + c4 * 8);
    async_b128(g, l);
  }
}

// One K=32 step for a 32x64 per-wave strip: four B fragments feed eight
// WMMAs (two A fragments).  sched_group_barrier pins the schedule to
// "8 DS reads, then 8 WMMAs" so all B fragments stay live in distinct
// registers: the in-order DS returns mean only partial dscnt waits are
// paid and LDS latency hides behind back-to-back WMMA issue.
static __device__ __forceinline__ void wmma_step8(const f16* lbuf, int lr, int hs,
                                                  v16h a0, v16h a1,
                                                  v8f acc[2][4]) {
  const v16h b0 = frag_ld(lbuf + (size_t)(lr)      * LDB_, hs);
  const v16h b1 = frag_ld(lbuf + (size_t)(16 + lr) * LDB_, hs);
  const v16h b2 = frag_ld(lbuf + (size_t)(32 + lr) * LDB_, hs);
  const v16h b3 = frag_ld(lbuf + (size_t)(48 + lr) * LDB_, hs);
  acc[0][0] = wmma16(a0, b0, acc[0][0]);
  acc[1][0] = wmma16(a1, b0, acc[1][0]);
  acc[0][1] = wmma16(a0, b1, acc[0][1]);
  acc[1][1] = wmma16(a1, b1, acc[1][1]);
  acc[0][2] = wmma16(a0, b2, acc[0][2]);
  acc[1][2] = wmma16(a1, b2, acc[1][2]);
  acc[0][3] = wmma16(a0, b3, acc[0][3]);
  acc[1][3] = wmma16(a1, b3, acc[1][3]);
  __builtin_amdgcn_sched_group_barrier(0x100, 8, 0);  // DS read x8
  __builtin_amdgcn_sched_group_barrier(0x008, 8, 0);  // MFMA/WMMA x8
}

// ---------------------------------------------------------------------------
// Kernel 1: RMSNorm over D=1024, cast to f16.  One block (256 thr) per token.
// ---------------------------------------------------------------------------
__global__ void k_rmsnorm(const float* __restrict__ x,
                          const float* __restrict__ scale,
                          f16* __restrict__ hbuf) {
  const int row  = blockIdx.x;                 // token 0..4095
  const int tid  = threadIdx.x;                // 0..255
  const float4 v = *(const float4*)(x + (size_t)row * D_ + tid * 4);

  float ss = v.x * v.x + v.y * v.y + v.z * v.z + v.w * v.w;
  #pragma unroll
  for (int off = 16; off >= 1; off >>= 1) ss += __shfl_xor(ss, off, 32);

  __shared__ float red[8];
  if ((tid & 31) == 0) red[tid >> 5] = ss;
  __syncthreads();
  float tot = 0.f;
  #pragma unroll
  for (int i = 0; i < 8; ++i) tot += red[i];

  const float r = rsqrtf(tot * (1.0f / D_) + 1e-6f);
  f16* out = hbuf + (size_t)row * D_ + tid * 4;
  const float4 sc = *(const float4*)(scale + tid * 4);
  out[0] = (f16)(v.x * r * sc.x);
  out[1] = (f16)(v.y * r * sc.y);
  out[2] = (f16)(v.z * r * sc.z);
  out[3] = (f16)(v.w * r * sc.w);
}

// ---------------------------------------------------------------------------
// Kernel 2: cast 4x [1024][1024] f32 weights to f16, transposed (N-major)
// so WMMA B-fragments load contiguously.  z = 0..2 -> wq/wk/wv, z = 3 -> wo.
// ---------------------------------------------------------------------------
__global__ void k_wcast(const float* __restrict__ wq,
                        const float* __restrict__ wk,
                        const float* __restrict__ wv,
                        const float* __restrict__ wo,
                        f16* __restrict__ wT,    // 3 x [HE][D]
                        f16* __restrict__ woT) { // [D][HE]
  __shared__ float tile[32][33];
  const int z = blockIdx.z;
  const float* src = (z == 0) ? wq : (z == 1) ? wk : (z == 2) ? wv : wo;
  f16* dst = (z < 3) ? (wT + (size_t)z * HE_ * D_) : woT;

  const int x = blockIdx.x * 32 + threadIdx.x;
  const int y = blockIdx.y * 32 + threadIdx.y;
  #pragma unroll
  for (int yy = 0; yy < 32; yy += 8)
    tile[threadIdx.y + yy][threadIdx.x] = src[(size_t)(y + yy) * 1024 + x];
  __syncthreads();
  const int x2 = blockIdx.y * 32 + threadIdx.x;
  const int y2 = blockIdx.x * 32 + threadIdx.y;
  #pragma unroll
  for (int yy = 0; yy < 32; yy += 8)
    dst[(size_t)(y2 + yy) * 1024 + x2] = (f16)tile[threadIdx.x][threadIdx.y + yy];
}

// ---------------------------------------------------------------------------
// Kernel 3: QKV GEMM.  C[4096 x 1024] = h[4096 x 1024] * W.
// Block = 128 thr (4 waves); wave computes a 32x64 strip (2 A frags x 4 B
// frags = 8 accumulators).  B tiles stream through double-buffered LDS via
// async copies (two explicit phases per iteration -> compile-time LDS
// addresses); A fragments register double-buffered.  launch_bounds(128,1)
// lifts the VGPR cap so the whole pipeline stays in registers (no spills).
// grid = (HE/64, NTOK/128, 3).
// ---------------------------------------------------------------------------
__global__ void __launch_bounds__(128, 1)
k_qkv(const f16* __restrict__ hbuf,
      const f16* __restrict__ wT,
      f16* __restrict__ qT, f16* __restrict__ kT,
      f16* __restrict__ vT) {
  __shared__ __align__(16) f16 buf0[64 * LDB_];   // 5 KB
  __shared__ __align__(16) f16 buf1[64 * LDB_];   // 5 KB
  const int tid  = threadIdx.x;
  const int lane = tid & 31;
  const int wave = tid >> 5;
  const int hs   = lane >> 4;
  const int lr   = lane & 15;
  const int m0   = blockIdx.y * 128 + wave * 32;  // token tile (32 rows)
  const int n0   = blockIdx.x * 64;               // (h,e) tile
  const f16* Wn = wT + (size_t)blockIdx.z * HE_ * D_ + (size_t)n0 * D_;
  const f16* Ar0 = hbuf + (size_t)(m0 + lr) * D_;
  const f16* Ar1 = hbuf + (size_t)(m0 + 16 + lr) * D_;

  stage_b64x32(Wn, buf0, tid);
  v16h a0 = frag_ld(Ar0, hs);
  v16h a1 = frag_ld(Ar1, hs);

  v8f acc[2][4] = {{{}, {}, {}, {}}, {{}, {}, {}, {}}};
  for (int k0 = 0; k0 < D_; k0 += 64) {
    // ---- phase A: consume buf0 (K = k0), prefetch K = k0+32 into buf1 ----
    wait_async0();
    __syncthreads();
    stage_b64x32(Wn + k0 + 32, buf1, tid);           // k0+32 < D_ always
    const v16h a0n = frag_ld(Ar0 + k0 + 32, hs);     // overlaps WMMAs
    const v16h a1n = frag_ld(Ar1 + k0 + 32, hs);
    wmma_step8(buf0, lr, hs, a0, a1, acc);

    // ---- phase B: consume buf1 (K = k0+32), prefetch K = k0+64 ----
    wait_async0();
    __syncthreads();
    const int k2 = (k0 + 64 < D_) ? (k0 + 64) : k0;  // clamp on last iter
    if (k0 + 64 < D_) stage_b64x32(Wn + k0 + 64, buf0, tid);
    a0 = frag_ld(Ar0 + k2, hs);
    a1 = frag_ld(Ar1 + k2, hs);
    wmma_step8(buf1, lr, hs, a0n, a1n, acc);
  }

  // C layout: lanes 0-15 -> rows r, lanes 16-31 -> rows r+8; col = lane%16.
  #pragma unroll
  for (int mi = 0; mi < 2; ++mi) {
    #pragma unroll
    for (int nt = 0; nt < 4; ++nt) {
      #pragma unroll
      for (int r = 0; r < 8; ++r) {
        const int row = m0 + mi * 16 + r + hs * 8;    // token
        const int n   = n0 + nt * 16 + lr;            // h*64+e
        const int b   = row >> 11, s = row & (S_ - 1);
        const int h   = n >> 6,  e = n & (DH_ - 1);
        const float v = acc[mi][nt][r];
        const size_t se = (((size_t)(b * H_ + h) * S_) + s) * DH_ + e;
        if (blockIdx.z == 0)      qT[se] = (f16)(v * 0.125f);   // Dh^-0.5
        else if (blockIdx.z == 1) kT[se] = (f16)v;
        else vT[(((size_t)(b * H_ + h) * DH_) + e) * S_ + s] = (f16)v;
      }
    }
  }
}

// ---------------------------------------------------------------------------
// Kernel 4: flash-style attention.  Block = 128 thr (4 waves); each wave
// owns one 16-query tile; grid = (S/64, B*H).  kv processed in steps of 32.
// V fragments are loaded before the softmax so their latency hides behind
// the exp/shuffle VALU work.
// ---------------------------------------------------------------------------
__global__ void __launch_bounds__(128, 1)
k_attn(const f16* __restrict__ qT, const f16* __restrict__ kT,
       const f16* __restrict__ vT, f16* __restrict__ att) {
  __shared__ __align__(16) f16 pT[4][16][32];   // per-wave P staging (1 KB)

  const int lane = threadIdx.x & 31;
  const int wave = threadIdx.x >> 5;
  const int hs   = lane >> 4;
  const int lr   = lane & 15;
  const int bh   = blockIdx.y;                  // b*H + h
  const int q0   = (blockIdx.x * 4 + wave) * 16;

  const f16* Qb = qT + (size_t)bh * S_ * DH_;
  const f16* Kb = kT + (size_t)bh * S_ * DH_;
  const f16* Vb = vT + (size_t)bh * DH_ * S_;

  const v16h qf0 = frag_ld(Qb + (size_t)(q0 + lr) * DH_ + 0,  hs);
  const v16h qf1 = frag_ld(Qb + (size_t)(q0 + lr) * DH_ + 32, hs);

  float mi[8], li[8];
  v8f ao[4] = {{}, {}, {}, {}};
  #pragma unroll
  for (int r = 0; r < 8; ++r) { mi[r] = -1e30f; li[r] = 0.f; }

  for (int kv0 = 0; kv0 < S_; kv0 += 32) {
    // ---- issue K and V fragment loads together; V used only after softmax
    const v16h kf00 = frag_ld(Kb + (size_t)(kv0 + lr)      * DH_ + 0,  hs);
    const v16h kf01 = frag_ld(Kb + (size_t)(kv0 + lr)      * DH_ + 32, hs);
    const v16h kf10 = frag_ld(Kb + (size_t)(kv0 + 16 + lr) * DH_ + 0,  hs);
    const v16h kf11 = frag_ld(Kb + (size_t)(kv0 + 16 + lr) * DH_ + 32, hs);
    v16h vf[4];
    #pragma unroll
    for (int et = 0; et < 4; ++et)
      vf[et] = frag_ld(Vb + (size_t)(et * 16 + lr) * S_ + kv0, hs);

    // ---- S = Q * K^T for two 16-column tiles (f32 accumulate) ----
    v8f s0 = {}, s1 = {};
    s0 = wmma16(qf0, kf00, s0);
    s0 = wmma16(qf1, kf01, s0);
    s1 = wmma16(qf0, kf10, s1);
    s1 = wmma16(qf1, kf11, s1);

    // ---- online softmax (f32).  Row r lives on one 16-lane half. ----
    float alpha[8];
    #pragma unroll
    for (int r = 0; r < 8; ++r) {
      float bmax = fmaxf(s0[r], s1[r]);
      #pragma unroll
      for (int off = 8; off >= 1; off >>= 1)
        bmax = fmaxf(bmax, __shfl_xor(bmax, off, 32));
      const float nm = fmaxf(mi[r], bmax);
      alpha[r] = __expf(mi[r] - nm);
      mi[r] = nm;
      const float p0 = __expf(s0[r] - nm);
      const float p1 = __expf(s1[r] - nm);
      s0[r] = p0; s1[r] = p1;
      float rs = p0 + p1;
      #pragma unroll
      for (int off = 8; off >= 1; off >>= 1) rs += __shfl_xor(rs, off, 32);
      li[r] = li[r] * alpha[r] + rs;
    }
    #pragma unroll
    for (int et = 0; et < 4; ++et)
      #pragma unroll
      for (int r = 0; r < 8; ++r) ao[et][r] *= alpha[r];

    // ---- restage P: C layout -> A layout via per-wave LDS tile ----
    #pragma unroll
    for (int r = 0; r < 8; ++r) {
      pT[wave][r + hs * 8][lr]      = (f16)s0[r];
      pT[wave][r + hs * 8][16 + lr] = (f16)s1[r];
    }
    asm volatile("s_wait_dscnt 0x0" ::: "memory");
    const v16h pf = frag_ld(&pT[wave][lr][0], hs);

    // ---- O += P * V (V stored [e][s] -> contiguous B-fragments) ----
    #pragma unroll
    for (int et = 0; et < 4; ++et) ao[et] = wmma16(pf, vf[et], ao[et]);
  }

  // ---- normalize and write att[token][h*64+e] (f16) ----
  const int b = bh >> 4, h = bh & (H_ - 1);
  #pragma unroll
  for (int et = 0; et < 4; ++et) {
    #pragma unroll
    for (int r = 0; r < 8; ++r) {
      const int row    = q0 + r + hs * 8;
      const size_t tok = (size_t)b * S_ + row;
      att[tok * HE_ + h * DH_ + et * 16 + lr] = (f16)(ao[et][r] / li[r]);
    }
  }
}

// ---------------------------------------------------------------------------
// Kernel 5: output GEMM.  out[4096 x 1024] = att * Wo + b_out (f32 out).
// Same two-phase async-LDS structure as k_qkv.  grid = (D/64, NTOK/128).
// ---------------------------------------------------------------------------
__global__ void __launch_bounds__(128, 1)
k_outproj(const f16* __restrict__ att,
          const f16* __restrict__ woT,
          const float* __restrict__ b_out,
          float* __restrict__ out) {
  __shared__ __align__(16) f16 buf0[64 * LDB_];
  __shared__ __align__(16) f16 buf1[64 * LDB_];
  const int tid  = threadIdx.x;
  const int lane = tid & 31;
  const int wave = tid >> 5;
  const int hs   = lane >> 4;
  const int lr   = lane & 15;
  const int m0   = blockIdx.y * 128 + wave * 32;  // token tile (32 rows)
  const int n0   = blockIdx.x * 64;               // d tile
  const f16* Wn  = woT + (size_t)n0 * HE_;
  const f16* Ar0 = att + (size_t)(m0 + lr) * HE_;
  const f16* Ar1 = att + (size_t)(m0 + 16 + lr) * HE_;

  stage_b64x32(Wn, buf0, tid);
  v16h a0 = frag_ld(Ar0, hs);
  v16h a1 = frag_ld(Ar1, hs);

  v8f acc[2][4] = {{{}, {}, {}, {}}, {{}, {}, {}, {}}};
  for (int k0 = 0; k0 < HE_; k0 += 64) {
    wait_async0();
    __syncthreads();
    stage_b64x32(Wn + k0 + 32, buf1, tid);
    const v16h a0n = frag_ld(Ar0 + k0 + 32, hs);
    const v16h a1n = frag_ld(Ar1 + k0 + 32, hs);
    wmma_step8(buf0, lr, hs, a0, a1, acc);

    wait_async0();
    __syncthreads();
    const int k2 = (k0 + 64 < HE_) ? (k0 + 64) : k0;
    if (k0 + 64 < HE_) stage_b64x32(Wn + k0 + 64, buf0, tid);
    a0 = frag_ld(Ar0 + k2, hs);
    a1 = frag_ld(Ar1 + k2, hs);
    wmma_step8(buf1, lr, hs, a0n, a1n, acc);
  }

  #pragma unroll
  for (int mi = 0; mi < 2; ++mi) {
    #pragma unroll
    for (int nt = 0; nt < 4; ++nt) {
      #pragma unroll
      for (int r = 0; r < 8; ++r) {
        const int row = m0 + mi * 16 + r + hs * 8;
        const int n   = n0 + nt * 16 + lr;
        out[(size_t)row * D_ + n] = acc[mi][nt][r] + b_out[n];
      }
    }
  }
}

// ---------------------------------------------------------------------------
// Host launcher.  Inputs: x, norm_scale, wq, wk, wv, w_out, b_out (all f32).
// ---------------------------------------------------------------------------
extern "C" void kernel_launch(void* const* d_in, const int* in_sizes, int n_in,
                              void* d_out, int out_size, void* d_ws, size_t ws_size,
                              hipStream_t stream) {
  const float* x     = (const float*)d_in[0];
  const float* nsc   = (const float*)d_in[1];
  const float* wq    = (const float*)d_in[2];
  const float* wk    = (const float*)d_in[3];
  const float* wv    = (const float*)d_in[4];
  const float* wo    = (const float*)d_in[5];
  const float* b_out = (const float*)d_in[6];
  float* out = (float*)d_out;

  // Workspace partition (f16 staging, 48 MB total).
  char* ws = (char*)d_ws;
  const size_t MB = 1024 * 1024;
  f16* hbuf = (f16*)(ws + 0);        //  8 MB : [4096][1024] normalized f16
  f16* wT   = (f16*)(ws + 8 * MB);   //  6 MB : wq/wk/wv transposed [HE][D]
  f16* woT  = (f16*)(ws + 14 * MB);  //  2 MB : w_out transposed [D][HE]
  f16* qT   = (f16*)(ws + 16 * MB);  //  8 MB : [b,h,s,e] (pre-scaled)
  f16* kT   = (f16*)(ws + 24 * MB);  //  8 MB : [b,h,s,e]
  f16* vT   = (f16*)(ws + 32 * MB);  //  8 MB : [b,h,e,s]
  f16* attb = (f16*)(ws + 40 * MB);  //  8 MB : [token][h*64+e]

  k_rmsnorm<<<NTOK, 256, 0, stream>>>(x, nsc, hbuf);
  k_wcast<<<dim3(32, 32, 4), dim3(32, 8), 0, stream>>>(wq, wk, wv, wo, wT, woT);
  k_qkv<<<dim3(HE_ / 64, NTOK / 128, 3), 128, 0, stream>>>(hbuf, wT, qT, kT, vT);
  k_attn<<<dim3(S_ / 64, B_ * H_), 128, 0, stream>>>(qT, kT, vT, attb);
  k_outproj<<<dim3(D_ / 64, NTOK / 128), 128, 0, stream>>>(attb, woT, b_out, out);
}
